// MultiHeadLatentAttention_77068893160063
// MI455X (gfx1250) — matrix operand order repaired and checked by reading, not compiled, not run on hardware
//
#include <hip/hip_runtime.h>
#include <hip/hip_bf16.h>
#include <math.h>

// ---------------------------------------------------------------------------
// MLA forward for MI455X (gfx1250, wave32).
// GEMMs: TDM (tensor_load_to_lds) double-buffered LDS staging + WMMA bf16.
// Attention: flash attention with WMMA for QK^T and P*V.
// ---------------------------------------------------------------------------

typedef __bf16 bf16_t;
typedef __bf16 v16bf __attribute__((ext_vector_type(16)));
typedef float  v8f   __attribute__((ext_vector_type(8)));
typedef unsigned int u32x4 __attribute__((ext_vector_type(4)));
typedef int          i32x4 __attribute__((ext_vector_type(4)));
typedef int          i32x8 __attribute__((ext_vector_type(8)));

union Frag {                // 32 bytes: one WMMA A/B operand per lane
    v16bf v;
    uint4 q[2];
};

__device__ __forceinline__ v8f zero8() {
    v8f v;
#pragma unroll
    for (int i = 0; i < 8; ++i) v[i] = 0.0f;
    return v;
}

__device__ __forceinline__ v8f wmma_bf16(const Frag& a, const Frag& b, v8f c) {
    return __builtin_amdgcn_wmma_f32_16x16x32_bf16(
        /*neg_a=*/false, a.v, /*neg_b=*/false, b.v,
        /*c_mod=*/(short)0, c, /*reuse_a=*/false, /*reuse_b=*/false);
}

// ---------------------------------------------------------------------------
// TDM: 2D tile load Global -> LDS via Tensor Data Mover (CDNA5 ISA).
// D# group0: count=1 | lds_addr[63:32] | global_addr[120:64] | type=2.
// D# group1: data_size=2B | tensor_dim0/1 | tile_dim0/1 | dim0 stride.
// tile_dim0 = contiguous K run (bytes multiple of 4), tile_dim1 = rows.
// ---------------------------------------------------------------------------
__device__ __forceinline__ void tdm_load_2d(unsigned ldsOff, const void* gptr,
                                            unsigned tileK, unsigned tileRows,
                                            unsigned tensorK, unsigned tensorRows,
                                            unsigned long long strideElems) {
    unsigned long long ga = (unsigned long long)gptr;
    u32x4 g0;
    g0.x = 1u;                                            // count=1 (valid user D#)
    g0.y = ldsOff;                                        // lds_addr (bytes)
    g0.z = (unsigned)(ga & 0xFFFFFFFFull);                // global_addr[31:0]
    g0.w = (unsigned)((ga >> 32) & 0x01FFFFFFull)         // global_addr[56:32]
         | (2u << 30);                                    // type = 2 ("image")
    i32x8 g1;
    g1[0] = (int)(1u << 16);                              // data_size=1 -> 2 bytes
    g1[1] = (int)((tensorK & 0xFFFFu) << 16);             // tensor_dim0[15:0]
    g1[2] = (int)((tensorK >> 16) | ((tensorRows & 0xFFFFu) << 16));
    g1[3] = (int)((tensorRows >> 16) | (tileK << 16));    // tile_dim0
    g1[4] = (int)tileRows;                                // tile_dim1 (tile_dim2=0)
    g1[5] = (int)(strideElems & 0xFFFFFFFFull);           // dim0 stride lo
    g1[6] = (int)((strideElems >> 32) & 0xFFFFull);       // dim0 stride hi
    g1[7] = 0;                                            // dim1 stride (2D: unused)
    i32x4 z4 = {0, 0, 0, 0};
#if defined(__clang_major__) && (__clang_major__ >= 23)
    i32x8 z8 = {0, 0, 0, 0, 0, 0, 0, 0};
    __builtin_amdgcn_tensor_load_to_lds(g0, g1, z4, z4, z8, 0);
#else
    __builtin_amdgcn_tensor_load_to_lds(g0, g1, z4, z4, 0);
#endif
}

// ---------------------------------------------------------------------------
// Elementwise fp32 -> bf16
// ---------------------------------------------------------------------------
__global__ __launch_bounds__(256) void k_f32_to_bf16(const float* __restrict__ in,
                                                     bf16_t* __restrict__ out, int n) {
    int i = blockIdx.x * blockDim.x + threadIdx.x;
    if (i < n) out[i] = (bf16_t)in[i];
}

// ---------------------------------------------------------------------------
// Transposing fp32 -> bf16 ([R][C] -> [C][R]); also used to build V^T.
// ---------------------------------------------------------------------------
__global__ __launch_bounds__(256) void k_transpose_cvt(const float* __restrict__ in,
                                                       bf16_t* __restrict__ out,
                                                       int R, int C) {
    int i = blockIdx.x * blockDim.x + threadIdx.x;
    if (i < R * C) {
        int r = i / C, c = i - r * C;
        out[(size_t)c * R + r] = (bf16_t)in[i];
    }
}

// ---------------------------------------------------------------------------
// RoPE on fp32 [S][Hn*128] -> bf16 same layout. One thread per (s,h,pair).
// ---------------------------------------------------------------------------
__global__ __launch_bounds__(256) void k_rope_cvt(const float* __restrict__ in,
                                                  const float* __restrict__ freqs,
                                                  bf16_t* __restrict__ out,
                                                  int Slen, int Hn) {
    int idx = blockIdx.x * blockDim.x + threadIdx.x;
    int total = Slen * Hn * 64;
    if (idx >= total) return;
    int i = idx & 63;
    int h = (idx >> 6) % Hn;
    int s = idx / (64 * Hn);
    const float* base = in + ((size_t)s * Hn + h) * 128 + 2 * i;
    float x0 = base[0], x1 = base[1];
    float cs = freqs[((size_t)s * 64 + i) * 2 + 0];
    float sn = freqs[((size_t)s * 64 + i) * 2 + 1];
    bf16_t* ob = out + ((size_t)s * Hn + h) * 128 + 2 * i;
    ob[0] = (bf16_t)(x0 * cs - x1 * sn);
    ob[1] = (bf16_t)(x0 * sn + x1 * cs);
}

// ---------------------------------------------------------------------------
// GEMM: C[M][N] (f32) = A[M][K] (bf16 row-major) * Bt[N][K] (bf16, B^T).
// Block = 8 waves (4 M x 2 N), 128x128 C tile. K consumed in 64-deep chunks,
// double-buffered in LDS (64 KB); wave 0 drives the Tensor Data Mover
// (tensor_load_to_lds + s_wait_tensorcnt), all waves consume via ds_load_b128
// and issue 16 WMMAs per chunk from registers (2x4 macro-tile per wave).
// ---------------------------------------------------------------------------
__global__ __launch_bounds__(256) void k_gemm_bf16_wmma(
    const bf16_t* __restrict__ A, const bf16_t* __restrict__ Bt,
    float* __restrict__ C, int M, int N, int K, int lda, int ldbt, int ldc) {
    __shared__ bf16_t sA[2][128 * 64];   // 32 KB: A panel  [row][k]
    __shared__ bf16_t sB[2][128 * 64];   // 32 KB: Bt panel [col][k]

    const int lane = threadIdx.x & 31;
    const int wave = threadIdx.x >> 5;
    const int nbn  = N >> 7;                     // 128-wide column blocks
    const int bm   = blockIdx.x / nbn;
    const int bn   = blockIdx.x - bm * nbn;
    const int wm   = wave >> 1;                  // 0..3: 32-row strip
    const int wn   = wave & 1;                   // 0..1: 64-col strip
    const int colL = lane & 15;
    const int hi   = (lane & 16) ? 1 : 0;
    const int row0 = bm * 128, col0 = bn * 128;

    v8f acc[2][4];
#pragma unroll
    for (int i = 0; i < 2; ++i)
#pragma unroll
        for (int j = 0; j < 4; ++j) acc[i][j] = zero8();

    // wave 0 issues both panel DMAs for chunk starting at kc into buffer buf
    auto issue = [&](int buf, int kc) {
        tdm_load_2d((unsigned)(size_t)&sA[buf][0], A + (size_t)row0 * lda + kc,
                    64, 128, (unsigned)K, (unsigned)M, (unsigned long long)lda);
        tdm_load_2d((unsigned)(size_t)&sB[buf][0], Bt + (size_t)col0 * ldbt + kc,
                    64, 128, (unsigned)K, (unsigned)N, (unsigned long long)ldbt);
    };

    if (wave == 0) {
        issue(0, 0);
        __builtin_amdgcn_s_wait_tensorcnt(0);
    }
    __syncthreads();

    for (int kc = 0; kc < K; kc += 64) {
        const int cur = (kc >> 6) & 1;
        if (wave == 0 && kc + 64 < K) issue(cur ^ 1, kc + 64);

        const bf16_t* a0 = &sA[cur][(wm * 32 + colL) * 64] + (hi ? 8 : 0);
        const bf16_t* a1 = a0 + 16 * 64;
        const bf16_t* bb = &sB[cur][(wn * 64 + colL) * 64] + (hi ? 8 : 0);
#pragma unroll
        for (int ks = 0; ks < 64; ks += 32) {
            Frag fa0, fa1, fb[4];
            fa0.q[0] = *(const uint4*)(a0 + ks);        // K kb..kb+7
            fa0.q[1] = *(const uint4*)(a0 + ks + 16);   // K kb+16..kb+23
            fa1.q[0] = *(const uint4*)(a1 + ks);
            fa1.q[1] = *(const uint4*)(a1 + ks + 16);
#pragma unroll
            for (int j = 0; j < 4; ++j) {
                const bf16_t* bp = bb + j * 16 * 64 + ks;
                fb[j].q[0] = *(const uint4*)(bp);       // K kb..kb+7
                fb[j].q[1] = *(const uint4*)(bp + 16);  // K kb+16..kb+23
            }
#pragma unroll
            for (int j = 0; j < 4; ++j) {
                acc[0][j] = wmma_bf16(fa0, fb[j], acc[0][j]);
                acc[1][j] = wmma_bf16(fa1, fb[j], acc[1][j]);
            }
        }
        if (wave == 0 && kc + 64 < K) __builtin_amdgcn_s_wait_tensorcnt(0);
        __syncthreads();
    }

#pragma unroll
    for (int i = 0; i < 2; ++i)
#pragma unroll
        for (int j = 0; j < 4; ++j)
#pragma unroll
            for (int r = 0; r < 8; ++r) {
                int row = row0 + wm * 32 + i * 16 + r + (hi ? 8 : 0);
                int col = col0 + wn * 64 + j * 16 + colL;
                C[(size_t)row * ldc + col] = acc[i][j][r];
            }
}

// ---------------------------------------------------------------------------
// Flash attention, causal, GQA 4:1. One wave per (head, 16-row Q tile).
// QK^T and P*V both via WMMA; P transposed from C-layout to A-layout via LDS.
// Qb: [S][32*128] bf16 (post-RoPE)   Kb: [S][8*128] bf16 (post-RoPE)
// Vt: [8*128][S]  bf16 (transposed)  Ob: [S][32*128] bf16
// ---------------------------------------------------------------------------
__global__ __launch_bounds__(256) void k_mla_attention(
    const bf16_t* __restrict__ Qb, const bf16_t* __restrict__ Kb,
    const bf16_t* __restrict__ Vt, bf16_t* __restrict__ Ob) {
    __shared__ bf16_t ldsP[8][16 * 32];              // 1 KB per wave

    const int lane = threadIdx.x & 31;
    const int wave = threadIdx.x >> 5;
    const int gw   = blockIdx.x * (blockDim.x >> 5) + wave;  // 0..4095
    const int h    = gw & 31;                        // query head
    const int qt   = gw >> 5;                        // q tile 0..127
    const int hkv  = h >> 2;                         // kv head (N_REP=4)
    const int q0   = qt * 16;
    const int colL = lane & 15;
    const int hi   = (lane & 16) ? 1 : 0;

    // Q fragments: 16x128 tile = 4 A-fragments over the head dim
    Frag qf[4];
    {
        const bf16_t* qrow = Qb + (size_t)(q0 + colL) * 4096 + h * 128 + (hi ? 8 : 0);
#pragma unroll
        for (int d = 0; d < 4; ++d) {
            qf[d].q[0] = *(const uint4*)(qrow + d * 32);
            qf[d].q[1] = *(const uint4*)(qrow + d * 32 + 16);
        }
    }

    v8f acc[8];
#pragma unroll
    for (int i = 0; i < 8; ++i) acc[i] = zero8();
    float mrow[8], lrow[8];
#pragma unroll
    for (int r = 0; r < 8; ++r) { mrow[r] = -1e30f; lrow[r] = 0.0f; }

    const float scale = 0.08838834764831845f;        // 1/sqrt(128)
    const int nkb = (q0 + 16 + 31) >> 5;             // 32-key blocks (causal)

    for (int kb = 0; kb < nkb; ++kb) {
        const int ks = kb * 32;

        // ---- scores: two 16x16 tiles, keys ks..ks+15 and ks+16..ks+31 ----
        v8f s0 = zero8(), s1 = zero8();
        const int dlo = hi ? 8 : 0;
        const bf16_t* k0p = Kb + (size_t)(ks + colL)      * 1024 + hkv * 128 + dlo;
        const bf16_t* k1p = Kb + (size_t)(ks + 16 + colL) * 1024 + hkv * 128 + dlo;
#pragma unroll
        for (int d = 0; d < 4; ++d) {
            Frag b0, b1;
            b0.q[0] = *(const uint4*)(k0p + d * 32);
            b0.q[1] = *(const uint4*)(k0p + d * 32 + 16);
            b1.q[0] = *(const uint4*)(k1p + d * 32);
            b1.q[1] = *(const uint4*)(k1p + d * 32 + 16);
            s0 = wmma_bf16(qf[d], b0, s0);
            s1 = wmma_bf16(qf[d], b1, s1);
        }

        // ---- online softmax (row = r + 8*hi; 16 lanes share one row) ----
#pragma unroll
        for (int r = 0; r < 8; ++r) {
            const int q = q0 + r + (hi ? 8 : 0);
            float v0 = s0[r] * scale, v1 = s1[r] * scale;
            if (ks + colL > q)      v0 = -1e30f;     // causal mask
            if (ks + 16 + colL > q) v1 = -1e30f;
            float t = fmaxf(v0, v1);
            t = fmaxf(t, __shfl_xor(t, 1, 32));
            t = fmaxf(t, __shfl_xor(t, 2, 32));
            t = fmaxf(t, __shfl_xor(t, 4, 32));
            t = fmaxf(t, __shfl_xor(t, 8, 32));
            const float mnew  = fmaxf(mrow[r], t);
            const float alpha = __expf(mrow[r] - mnew);
            const float p0 = __expf(v0 - mnew);
            const float p1 = __expf(v1 - mnew);
            float ps = p0 + p1;
            ps += __shfl_xor(ps, 1, 32);
            ps += __shfl_xor(ps, 2, 32);
            ps += __shfl_xor(ps, 4, 32);
            ps += __shfl_xor(ps, 8, 32);
            lrow[r] = lrow[r] * alpha + ps;
            mrow[r] = mnew;
#pragma unroll
            for (int n = 0; n < 8; ++n) acc[n][r] *= alpha;
            // stage P row-major in LDS for the A-layout reload
            const int row = r + (hi ? 8 : 0);
            ldsP[wave][row * 32 + colL]      = (bf16_t)p0;
            ldsP[wave][row * 32 + 16 + colL] = (bf16_t)p1;
        }

        // ---- reload P as A-fragment (16x32) and do P*V over 8 n-tiles ----
        Frag pf;
        {
            const bf16_t* pp = &ldsP[wave][colL * 32 + (hi ? 8 : 0)];
            pf.q[0] = *(const uint4*)(pp);
            pf.q[1] = *(const uint4*)(pp + 16);
        }
#pragma unroll
        for (int n = 0; n < 8; ++n) {
            Frag vf;
            const bf16_t* vp = Vt + (size_t)(hkv * 128 + n * 16 + colL) * 2048
                                  + ks + (hi ? 8 : 0);
            vf.q[0] = *(const uint4*)(vp);
            vf.q[1] = *(const uint4*)(vp + 16);
            acc[n] = wmma_bf16(pf, vf, acc[n]);
        }
    }

    // ---- normalize and write bf16 output ----
#pragma unroll
    for (int r = 0; r < 8; ++r) {
        const int q  = q0 + r + (hi ? 8 : 0);
        const float inv = 1.0f / lrow[r];
#pragma unroll
        for (int n = 0; n < 8; ++n)
            Ob[(size_t)q * 4096 + h * 128 + n * 16 + colL] = (bf16_t)(acc[n][r] * inv);
    }
}

// ---------------------------------------------------------------------------
// Host-side orchestration
// ---------------------------------------------------------------------------
extern "C" void kernel_launch(void* const* d_in, const int* in_sizes, int n_in,
                              void* d_out, int out_size, void* d_ws, size_t ws_size,
                              hipStream_t stream) {
    (void)in_sizes; (void)n_in; (void)out_size; (void)ws_size;
    const int S = 2048, D = 4096, RANK = 512, KVW = 1024, QW = 4096;

    const float* x    = (const float*)d_in[0];
    const float* fr   = (const float*)d_in[1];
    const float* wdkv = (const float*)d_in[2];
    const float* wuk  = (const float*)d_in[3];
    const float* wuv  = (const float*)d_in[4];
    const float* wq   = (const float*)d_in[5];
    const float* wo   = (const float*)d_in[6];

    float* out  = (float*)d_out;                 // [S][D]
    float* ckvf = out + (size_t)S * D;           // [S][RANK] (second tuple output)

    char* ws = (char*)d_ws;
    const size_t MBy = 1ull << 20;
    bf16_t* xb    = (bf16_t*)(ws + 0);           // 16 MB  x bf16
    bf16_t* wdkvt = (bf16_t*)(ws + 16 * MBy);    //  4 MB  W_DKV^T [512][4096]
    bf16_t* wukt  = (bf16_t*)(ws + 20 * MBy);    //  1 MB  W_UK^T  [1024][512]
    bf16_t* wuvt  = (bf16_t*)(ws + 21 * MBy);    //  1 MB  W_UV^T  [1024][512]
    bf16_t* wqt   = (bf16_t*)(ws + 22 * MBy);    // 32 MB  Wq^T    [4096][4096]
    bf16_t* wot   = (bf16_t*)(ws + 54 * MBy);    // 32 MB  Wo^T    [4096][4096]
    bf16_t* ckvb  = (bf16_t*)(ws + 86 * MBy);    //  2 MB  c_KV bf16
    float*  kf    = (float*) (ws + 88 * MBy);    //  8 MB  K pre-rope f32
    float*  vf    = (float*) (ws + 96 * MBy);    //  8 MB  V f32
    float*  qf    = (float*) (ws + 104 * MBy);   // 32 MB  Q pre-rope f32
    bf16_t* kbp   = (bf16_t*)(ws + 136 * MBy);   //  4 MB  K bf16 (post-rope)
    bf16_t* vtb   = (bf16_t*)(ws + 140 * MBy);   //  4 MB  V^T bf16 [1024][2048]
    bf16_t* qb    = (bf16_t*)(ws + 144 * MBy);   // 16 MB  Q bf16 (post-rope)
    bf16_t* attnb = (bf16_t*)(ws + 88 * MBy);    // 16 MB  attn out (aliases kf/vf)

    dim3 blk(256);
    auto g = [](long long n) { return dim3((unsigned)((n + 255) / 256)); };
    // GEMM block tile: 128 (M) x 128 (N), 8 waves per block
    auto gemm_grid = [](int M, int N) { return dim3((unsigned)((M / 128) * (N / 128))); };

    // 1) bf16 conversions / weight transposes
    k_f32_to_bf16 <<<g((long long)S * D), blk, 0, stream>>>(x, xb, S * D);
    k_transpose_cvt<<<g((long long)D * RANK), blk, 0, stream>>>(wdkv, wdkvt, D, RANK);
    k_transpose_cvt<<<g((long long)RANK * KVW), blk, 0, stream>>>(wuk, wukt, RANK, KVW);
    k_transpose_cvt<<<g((long long)RANK * KVW), blk, 0, stream>>>(wuv, wuvt, RANK, KVW);
    k_transpose_cvt<<<g((long long)D * QW), blk, 0, stream>>>(wq, wqt, D, QW);
    k_transpose_cvt<<<g((long long)QW * D), blk, 0, stream>>>(wo, wot, QW, D);

    // 2) c_KV = x @ W_DKV  (f32 result straight into d_out segment 2)
    k_gemm_bf16_wmma<<<gemm_grid(S, RANK), blk, 0, stream>>>(
        xb, wdkvt, ckvf, S, RANK, D, D, D, RANK);
    k_f32_to_bf16<<<g((long long)S * RANK), blk, 0, stream>>>(ckvf, ckvb, S * RANK);

    // 3) K, V, Q projections
    k_gemm_bf16_wmma<<<gemm_grid(S, KVW), blk, 0, stream>>>(
        ckvb, wukt, kf, S, KVW, RANK, RANK, RANK, KVW);
    k_gemm_bf16_wmma<<<gemm_grid(S, KVW), blk, 0, stream>>>(
        ckvb, wuvt, vf, S, KVW, RANK, RANK, RANK, KVW);
    k_gemm_bf16_wmma<<<gemm_grid(S, QW), blk, 0, stream>>>(
        xb, wqt, qf, S, QW, D, D, D, QW);

    // 4) RoPE (Q: 32 heads, K: 8 heads) + V transpose, all -> bf16
    k_rope_cvt<<<g((long long)S * 32 * 64), blk, 0, stream>>>(qf, fr, qb, S, 32);
    k_rope_cvt<<<g((long long)S * 8 * 64), blk, 0, stream>>>(kf, fr, kbp, S, 8);
    k_transpose_cvt<<<g((long long)S * KVW), blk, 0, stream>>>(vf, vtb, S, KVW);

    // 5) causal flash attention: 32 heads x 128 q-tiles = 4096 waves
    k_mla_attention<<<dim3(512), blk, 0, stream>>>(qb, kbp, vtb, attnb);

    // 6) out = attn @ Wo  (f32 into d_out segment 1)
    k_gemm_bf16_wmma<<<gemm_grid(S, D), blk, 0, stream>>>(
        attnb, wot, out, S, D, QW, QW, QW, D);
}
